// CaNetConv_12970801234187
// MI455X (gfx1250) — matrix operand order repaired and checked by reading, not compiled, not run on hardware
//
#include <hip/hip_runtime.h>
#include <hip/hip_bf16.h>

#define KH 4
#define DD 64
#define GAT_EPS 1e-8f

typedef __attribute__((ext_vector_type(16))) __bf16 v16bf;
typedef __attribute__((ext_vector_type(8)))  float  v8f;

// ---------- helpers ----------
static __device__ __forceinline__ unsigned short f32_to_bf16_rne(float f) {
  unsigned u = __float_as_uint(f);
  unsigned r = u + 0x7FFFu + ((u >> 16) & 1u);
  return (unsigned short)(r >> 16);
}
static __device__ __forceinline__ float bf16bits_to_f32(unsigned short s) {
  return __uint_as_float(((unsigned)s) << 16);
}
static __device__ __forceinline__ __bf16 bits_to_bf16(unsigned short s) {
  __bf16 b; __builtin_memcpy(&b, &s, 2); return b;
}
static __device__ __forceinline__ void split_bf16(float f, __bf16& hi, __bf16& lo) {
  unsigned short hb = f32_to_bf16_rne(f);
  hi = bits_to_bf16(hb);
  float resid = f - bf16bits_to_f32(hb);
  lo = bits_to_bf16(f32_to_bf16_rne(resid));
}
// non-returning f32 atomic add -> native global_atomic_add_f32 (L2 atomic unit)
static __device__ __forceinline__ void atomic_add_f32_nr(float* p, float v) {
  asm volatile("global_atomic_add_f32 %0, %1, off" :: "v"(p), "v"(v) : "memory");
}

// ---------- kernel 0: zero num / alpha / denom region ----------
__global__ __launch_bounds__(256) void gat_zero(float* p, long long n) {
  long long i = (long long)blockIdx.x * blockDim.x + threadIdx.x;
  if (i < n) p[i] = 0.0f;
}

// ---------- kernel 1: h = x @ W[k] (split-bf16 WMMA) + alpha_dst/src ----------
// grid: N/16 blocks of 128 threads (4 waves, wave = head). Each wave does a
// 16x64 tile: 4 output 16x16 tiles, K=64 in two 32-chunks, 3 WMMAs per chunk.
__global__ __launch_bounds__(128) void gat_gemm_alpha(
    const float* __restrict__ x, const float* __restrict__ w,
    const float* __restrict__ avec,
    float* __restrict__ h, float* __restrict__ alpha_d, float* __restrict__ alpha_s,
    int N)
{
  const int head  = threadIdx.x >> 5;
  const int lane  = threadIdx.x & 31;
  const int rlow  = lane & 15;
  const int rhigh = lane >> 4;
  const int m0    = blockIdx.x * 16;

  const float* wk = w + (size_t)head * DD * DD;

  v8f acc[4];
  #pragma unroll
  for (int ot = 0; ot < 4; ++ot)
    #pragma unroll
    for (int i = 0; i < 8; ++i) acc[ot][i] = 0.0f;

  #pragma unroll
  for (int kk = 0; kk < 2; ++kk) {
    // ---- A fragment: row m = m0+rlow; elem e: K = kk*32 + rhigh*16 + e
    const int m = m0 + rlow;
    const float* xr = x + (size_t)m * DD + kk * 32 + rhigh * 16;
    float av[16];
    {
      float4 g0a = *(const float4*)(xr + 0);
      float4 g0b = *(const float4*)(xr + 4);
      float4 g1a = *(const float4*)(xr + 8);
      float4 g1b = *(const float4*)(xr + 12);
      av[0]=g0a.x; av[1]=g0a.y; av[2]=g0a.z; av[3]=g0a.w;
      av[4]=g0b.x; av[5]=g0b.y; av[6]=g0b.z; av[7]=g0b.w;
      av[8]=g1a.x; av[9]=g1a.y; av[10]=g1a.z; av[11]=g1a.w;
      av[12]=g1b.x; av[13]=g1b.y; av[14]=g1b.z; av[15]=g1b.w;
    }
    v16bf a_hi, a_lo;
    #pragma unroll
    for (int e = 0; e < 16; ++e) {
      __bf16 hi, lo; split_bf16(av[e], hi, lo);
      a_hi[e] = hi; a_lo[e] = lo;
    }

    #pragma unroll
    for (int ot = 0; ot < 4; ++ot) {
      // ---- B fragment: col n = ot*16+rlow; elem e: K = kk*32 + rhigh*16 + e
      const float* wb = wk + (size_t)(kk * 32 + rhigh * 16) * DD + ot * 16 + rlow;
      v16bf b_hi, b_lo;
      #pragma unroll
      for (int e = 0; e < 16; ++e) {
        __bf16 hi, lo; split_bf16(wb[(size_t)e * DD], hi, lo);
        b_hi[e] = hi; b_lo[e] = lo;
      }
      acc[ot] = __builtin_amdgcn_wmma_f32_16x16x32_bf16(
          false, a_hi, false, b_hi, (short)0, acc[ot], false, false);
      acc[ot] = __builtin_amdgcn_wmma_f32_16x16x32_bf16(
          false, a_lo, false, b_hi, (short)0, acc[ot], false, false);
      acc[ot] = __builtin_amdgcn_wmma_f32_16x16x32_bf16(
          false, a_hi, false, b_lo, (short)0, acc[ot], false, false);
    }
  }

  // ---- store h; accumulate alpha partials (C/D layout: reg r -> row r+8*rhigh, col = lane&15)
  float pd[8], ps[8];
  #pragma unroll
  for (int r = 0; r < 8; ++r) { pd[r] = 0.0f; ps[r] = 0.0f; }

  float* hk = h + (size_t)head * N * DD;
  #pragma unroll
  for (int ot = 0; ot < 4; ++ot) {
    const float cd = avec[head * 2 * DD + ot * 16 + rlow];
    const float cs = avec[head * 2 * DD + DD + ot * 16 + rlow];
    #pragma unroll
    for (int r = 0; r < 8; ++r) {
      const int m = m0 + r + 8 * rhigh;
      const float v = acc[ot][r];
      hk[(size_t)m * DD + ot * 16 + rlow] = v;
      pd[r] += v * cd;
      ps[r] += v * cs;
    }
  }
  // reduce across the 16 lanes holding one row group
  #pragma unroll
  for (int r = 0; r < 8; ++r) {
    #pragma unroll
    for (int s = 1; s < 16; s <<= 1) {
      pd[r] += __shfl_xor(pd[r], s, 16);
      ps[r] += __shfl_xor(ps[r], s, 16);
    }
  }
  if (rlow == 0) {
    #pragma unroll
    for (int r = 0; r < 8; ++r) {
      const int m = m0 + r + 8 * rhigh;
      alpha_d[(size_t)head * N + m] = pd[r];
      alpha_s[(size_t)head * N + m] = ps[r];
    }
  }
}

// ---------- kernel 2: edge scatter ----------
// 16 lanes per (edge, head); each lane handles 4 consecutive dims (float4 read,
// 4 native f32 atomics into num). No softmax-max pass (cancels in num/denom).
__global__ __launch_bounds__(256) void gat_edge(
    const int* __restrict__ adj,
    const float* __restrict__ alpha_d, const float* __restrict__ alpha_s,
    const float* __restrict__ h,
    float* __restrict__ num, float* __restrict__ denom,
    int N, int E)
{
  const long long gid = (long long)blockIdx.x * 256 + threadIdx.x;
  const long long pair = gid >> 4;
  if (pair >= (long long)E * KH) return;
  const int oo   = ((int)gid & 15) * 4;
  const int edge = (int)(pair >> 2);
  const int head = (int)pair & 3;

  const int dstn = adj[edge];
  const int srcn = adj[E + edge];

  float lg = alpha_d[(size_t)head * N + dstn] + alpha_s[(size_t)head * N + srcn];
  lg = (lg > 0.0f) ? lg : 0.01f * lg;      // leaky_relu(0.01)
  const float wgt = expf(lg);

  const float4 hv = *(const float4*)(h + ((size_t)head * N + srcn) * DD + oo);
  float* np = num + ((size_t)head * N + dstn) * DD + oo;
  atomic_add_f32_nr(np + 0, wgt * hv.x);
  atomic_add_f32_nr(np + 1, wgt * hv.y);
  atomic_add_f32_nr(np + 2, wgt * hv.z);
  atomic_add_f32_nr(np + 3, wgt * hv.w);
  if (oo == 0)
    atomic_add_f32_nr(denom + (size_t)head * N + dstn, wgt);
}

// ---------- kernel 3: finalize ----------
__global__ __launch_bounds__(256) void gat_finalize(
    const float* __restrict__ x, const float* __restrict__ ein,
    const float* __restrict__ num, const float* __restrict__ denom,
    float* __restrict__ out, int N)
{
  const int gid = blockIdx.x * 256 + threadIdx.x;
  if (gid >= N * DD) return;
  const int n = gid >> 6;
  const int o = gid & 63;
  float acc = x[gid];
  #pragma unroll
  for (int k = 0; k < KH; ++k) {
    const float d = denom[(size_t)k * N + n] + GAT_EPS;
    acc += ein[n * KH + k] * num[((size_t)k * N + n) * DD + o] / d;
  }
  out[gid] = acc;
}

// ---------- launcher ----------
extern "C" void kernel_launch(void* const* d_in, const int* in_sizes, int n_in,
                              void* d_out, int out_size, void* d_ws, size_t ws_size,
                              hipStream_t stream) {
  const float* x    = (const float*)d_in[0];
  const int*   adj  = (const int*)d_in[1];
  const float* ein  = (const float*)d_in[2];
  const float* w    = (const float*)d_in[3];
  const float* avec = (const float*)d_in[4];
  float* out = (float*)d_out;

  const int N = in_sizes[0] / DD;   // 50000
  const int E = in_sizes[1] / 2;    // 850000

  float* ws = (float*)d_ws;
  const size_t hN = (size_t)KH * N * DD;
  float* h       = ws;                         // 4*N*64
  float* num     = ws + hN;                    // 4*N*64
  float* alpha_d = ws + 2 * hN;                // 4*N
  float* alpha_s = alpha_d + (size_t)KH * N;   // 4*N
  float* denom   = alpha_s + (size_t)KH * N;   // 4*N

  // zero num .. denom (contiguous: num, alpha_d, alpha_s, denom; alphas get
  // overwritten by the gemm kernel before use — zeroing them is harmless)
  const long long zeroN = (long long)hN + 3LL * KH * N;
  {
    int zb = (int)((zeroN + 255) / 256);
    gat_zero<<<zb, 256, 0, stream>>>(num, zeroN);
  }

  gat_gemm_alpha<<<N / 16, 128, 0, stream>>>(x, w, avec, h, alpha_d, alpha_s, N);

  {
    const long long ethreads = (long long)E * KH * 16;
    int eb = (int)((ethreads + 255) / 256);
    gat_edge<<<eb, 256, 0, stream>>>(adj, alpha_d, alpha_s, h, num, denom, N, E);
  }

  gat_finalize<<<(N * DD + 255) / 256, 256, 0, stream>>>(x, ein, num, denom, out, N);
}
